// FastTransformer_54082228191391
// MI455X (gfx1250) — matrix operand order repaired and checked by reading, not compiled, not run on hardware
//
#include <hip/hip_runtime.h>
#include <cstdint>
#include <cstddef>

// ---------------- problem constants ----------------
#define LYR   4
#define BB    32
#define NN    512
#define DIMD  512
#define HH    8
#define DHD   64
#define INNR  512
#define FFD   2048
#define OUTD  64
#define MNROW (BB * NN)   // 16384

typedef unsigned short u16;
typedef __bf16 bf16_t;
typedef bf16_t v16bf __attribute__((ext_vector_type(16)));
typedef float  v8f   __attribute__((ext_vector_type(8)));

static __device__ __forceinline__ u16 f2bf(float f) {
    unsigned u = __builtin_bit_cast(unsigned, f);
    unsigned r = u + 0x7FFFu + ((u >> 16) & 1u);   // round-to-nearest-even
    return (u16)(r >> 16);
}

// CDNA5 async global->LDS copy, 16B per lane, tracked by ASYNCcnt.
static __device__ __forceinline__ void async_cp16(unsigned lds_off, const void* gaddr) {
    asm volatile("global_load_async_to_lds_b128 %0, %1, off"
                 :: "v"(lds_off), "v"(gaddr) : "memory");
}

// ---------------- elementwise converts ----------------
__global__ void f32_to_bf16_kernel(const float* __restrict__ in, u16* __restrict__ out, size_t n) {
    size_t i = (size_t)blockIdx.x * blockDim.x + threadIdx.x;
    if (i < n) out[i] = f2bf(in[i]);
}

// W [K,N] fp32 -> Wt [N,K] bf16
__global__ void transpose_bf16_kernel(const float* __restrict__ in, u16* __restrict__ out, int K, int N) {
    size_t i = (size_t)blockIdx.x * blockDim.x + threadIdx.x;
    if (i < (size_t)K * N) {
        int n = (int)(i / K), k = (int)(i % K);
        out[i] = f2bf(in[(size_t)k * N + n]);
    }
}

// Expand per-head Wr[DH,DH] into block-diagonal WrExp^T [INNER,INNER] bf16 + brExp[INNER]
__global__ void build_wrexp_kernel(const float* __restrict__ Wr, const float* __restrict__ br,
                                   u16* __restrict__ outT, float* __restrict__ brExp) {
    size_t i = (size_t)blockIdx.x * blockDim.x + threadIdx.x;
    if (i < (size_t)INNR * INNR) {
        int n = (int)(i / INNR), k = (int)(i % INNR);
        float v = ((n >> 6) == (k >> 6)) ? Wr[(size_t)(k & 63) * DHD + (n & 63)] : 0.0f;
        outT[i] = f2bf(v);
    }
    if (i < INNR) brExp[i] = br[i & 63];
}

// u = v * global_k (broadcast over n), written directly as bf16
__global__ void gate_kernel(const float* __restrict__ v, const float* __restrict__ gk,
                            u16* __restrict__ u, size_t total) {
    size_t i = (size_t)blockIdx.x * blockDim.x + threadIdx.x;
    if (i < total) {
        int inner = (int)(i & (INNR - 1));          // h*64+d
        size_t bn = i >> 9;                          // /512
        int b = (int)(bn >> 9);                      // /N
        u[i] = f2bf(v[i] * gk[(size_t)b * INNR + inner]);
    }
}

// ---------------- fused global q/k pooling (one block per (b,h)) ----------------
__global__ __launch_bounds__(256) void pool_kernel(const float* __restrict__ q, const float* __restrict__ k,
                                                   const float* __restrict__ wql, const float* __restrict__ wkl,
                                                   float* __restrict__ gk) {
    const int bh = blockIdx.x;              // b*H + h
    const int b = bh >> 3, h = bh & 7;
    const int tid = threadIdx.x;
    __shared__ float lg[NN];
    __shared__ float red[256];
    __shared__ float gq[DHD];
    __shared__ float wl[DHD];
    const size_t base = (size_t)b * NN * INNR + (size_t)h * DHD;   // row stride INNER

    if (tid < DHD) wl[tid] = wql[tid];
    __syncthreads();
    for (int n = tid; n < NN; n += 256) {
        const float* qp = q + base + (size_t)n * INNR;
        float s = 0.f;
        #pragma unroll 8
        for (int d = 0; d < DHD; ++d) s += qp[d] * wl[d];
        lg[n] = s * 0.125f;                 // SCALE = DH^-0.5
    }
    __syncthreads();
    // softmax #1
    float m = fmaxf(lg[tid], lg[tid + 256]); red[tid] = m; __syncthreads();
    for (int st = 128; st > 0; st >>= 1) { if (tid < st) red[tid] = fmaxf(red[tid], red[tid + st]); __syncthreads(); }
    m = red[0]; __syncthreads();
    float acc = 0.f;
    for (int n = tid; n < NN; n += 256) { float e = __expf(lg[n] - m); lg[n] = e; acc += e; }
    red[tid] = acc; __syncthreads();
    for (int st = 128; st > 0; st >>= 1) { if (tid < st) red[tid] += red[tid + st]; __syncthreads(); }
    float inv = 1.f / red[0]; __syncthreads();
    // global_q
    if (tid < DHD) {
        float s = 0.f;
        for (int n = 0; n < NN; ++n) s += lg[n] * q[base + (size_t)n * INNR + tid];
        gq[tid] = s * inv;
    }
    __syncthreads();
    if (tid < DHD) wl[tid] = gq[tid] * wkl[tid];
    __syncthreads();
    for (int n = tid; n < NN; n += 256) {
        const float* kp = k + base + (size_t)n * INNR;
        float s = 0.f;
        #pragma unroll 8
        for (int d = 0; d < DHD; ++d) s += kp[d] * wl[d];
        lg[n] = s * 0.125f;
    }
    __syncthreads();
    // softmax #2
    m = fmaxf(lg[tid], lg[tid + 256]); red[tid] = m; __syncthreads();
    for (int st = 128; st > 0; st >>= 1) { if (tid < st) red[tid] = fmaxf(red[tid], red[tid + st]); __syncthreads(); }
    m = red[0]; __syncthreads();
    acc = 0.f;
    for (int n = tid; n < NN; n += 256) { float e = __expf(lg[n] - m); lg[n] = e; acc += e; }
    red[tid] = acc; __syncthreads();
    for (int st = 128; st > 0; st >>= 1) { if (tid < st) red[tid] += red[tid + st]; __syncthreads(); }
    inv = 1.f / red[0]; __syncthreads();
    if (tid < DHD) {
        float s = 0.f;
        for (int n = 0; n < NN; ++n) s += lg[n] * k[base + (size_t)n * INNR + tid];
        gk[(size_t)bh * DHD + tid] = s * inv;
    }
}

// ---------------- layernorm (row per block) -> bf16 ----------------
__global__ __launch_bounds__(256) void layernorm_kernel(const float* __restrict__ x, const float* __restrict__ g,
                                                        const float* __restrict__ bta, u16* __restrict__ out) {
    const int row = blockIdx.x, tid = threadIdx.x;
    __shared__ float red[256];
    const float* xr = x + (size_t)row * DIMD;
    float a = xr[tid], b2 = xr[tid + 256];
    red[tid] = a + b2; __syncthreads();
    for (int st = 128; st > 0; st >>= 1) { if (tid < st) red[tid] += red[tid + st]; __syncthreads(); }
    float mu = red[0] * (1.f / DIMD); __syncthreads();
    float d1 = a - mu, d2 = b2 - mu;
    red[tid] = d1 * d1 + d2 * d2; __syncthreads();
    for (int st = 128; st > 0; st >>= 1) { if (tid < st) red[tid] += red[tid + st]; __syncthreads(); }
    float rstd = rsqrtf(red[0] * (1.f / DIMD) + 1e-5f);
    out[(size_t)row * DIMD + tid]       = f2bf(d1 * rstd * g[tid] + bta[tid]);
    out[(size_t)row * DIMD + tid + 256] = f2bf(d2 * rstd * g[tid + 256] + bta[tid + 256]);
}

// ---------------- WMMA bf16 GEMM: C[M,N] = A[M,K] @ Bt[N,K]^T + bias, fused epilogue ----
// Async-to-LDS double-buffered staging, BKK=64 (two 32-K WMMA slices per stage).
// 8 waves as 4(M) x 2(N); wave tile 32 x (BN_T/2). Steady-state loop is branch-free.
// epi: 0=none, 1=abs, 2=gelu(exact), 3=+res[m,n]
#define BM  128
#define BKK 64

template <int BN_T>
__global__ __launch_bounds__(256) void gemm_bf16_kernel(
    const u16* __restrict__ A, const u16* __restrict__ Bt,
    const float* __restrict__ bias, const float* __restrict__ res,
    float* __restrict__ outF, u16* __restrict__ outH,
    int M, int N, int K, int epi) {
    constexpr int NFRAG = BN_T / 32;                  // 16-col frags per wave in N
    constexpr int NPEND = (BN_T == 128) ? 8 : 6;      // async ops per thread per stage
    __shared__ u16 lA[2][BM * BKK];
    __shared__ u16 lB[2][BN_T * BKK];
    const int tid  = threadIdx.x;
    const int lane = tid & 31, wave = tid >> 5;
    const int wm = wave >> 1, wn = wave & 1;
    const int hlf = lane >> 4, l15 = lane & 15;
    const int m0 = blockIdx.y * BM, n0 = blockIdx.x * BN_T;

    v8f c[2][NFRAG];
    #pragma unroll
    for (int im = 0; im < 2; ++im)
        #pragma unroll
        for (int in = 0; in < NFRAG; ++in) c[im][in] = v8f{};

    // staging geometry: each thread owns one contiguous run per tile
    const int arow = tid >> 1, aseg = tid & 1;                             // A: 64B/thread
    const int brow = (BN_T == 128) ? (tid >> 1) : (tid >> 2);
    const int bseg = (BN_T == 128) ? (tid & 1) : (tid & 3);
    const u16* gA = A  + (size_t)(m0 + arow) * K + aseg * 32;
    const u16* gB = Bt + (size_t)(n0 + brow) * K + bseg * ((BN_T == 128) ? 32 : 16);

    const unsigned lA_base = (unsigned)(uintptr_t)&lA[0][0];
    const unsigned lB_base = (unsigned)(uintptr_t)&lB[0][0];
    const unsigned laOff = (unsigned)((arow * BKK + aseg * 32) * 2);
    const unsigned lbOff = (unsigned)((brow * BKK + bseg * ((BN_T == 128) ? 32 : 16)) * 2);

    auto stage = [&](int buf, const u16* pa, const u16* pb) {
        unsigned la = lA_base + (unsigned)buf * (BM * BKK * 2) + laOff;
        async_cp16(la,      pa);
        async_cp16(la + 16, pa + 8);
        async_cp16(la + 32, pa + 16);
        async_cp16(la + 48, pa + 24);
        unsigned lb = lB_base + (unsigned)buf * (BN_T * BKK * 2) + lbOff;
        if constexpr (BN_T == 128) {
            async_cp16(lb,      pb);
            async_cp16(lb + 16, pb + 8);
            async_cp16(lb + 32, pb + 16);
            async_cp16(lb + 48, pb + 24);
        } else {
            async_cp16(lb,      pb);
            async_cp16(lb + 16, pb + 8);
        }
    };

    auto compute = [&](int buf) {
        const u16* la = &lA[buf][0];
        const u16* lb = &lB[buf][0];
        #pragma unroll
        for (int s = 0; s < 2; ++s) {                 // two 32-K slices
            union Frag { v16bf v; uint4 q[2]; } fa[2], fb[NFRAG];
            // A fragment: 16x32 bf16; lane(hlf,m): K chunks [hlf*8,+8) and [16+hlf*8,+8)
            #pragma unroll
            for (int im = 0; im < 2; ++im) {
                int r = wm * 32 + im * 16 + l15;
                fa[im].q[0] = *(const uint4*)(la + r * BKK + s * 32 + hlf * 8);
                fa[im].q[1] = *(const uint4*)(la + r * BKK + s * 32 + 16 + hlf * 8);
            }
            // B fragment: 32x16 bf16; lane(hlf,n): column n, K chunks [hlf*8,+8) and [16+hlf*8,+8)
            #pragma unroll
            for (int in = 0; in < NFRAG; ++in) {
                int cI = wn * (NFRAG * 16) + in * 16 + l15;
                fb[in].q[0] = *(const uint4*)(lb + cI * BKK + s * 32 + hlf * 8);
                fb[in].q[1] = *(const uint4*)(lb + cI * BKK + s * 32 + 16 + hlf * 8);
            }
            #pragma unroll
            for (int im = 0; im < 2; ++im)
                #pragma unroll
                for (int in = 0; in < NFRAG; ++in)
                    c[im][in] = __builtin_amdgcn_wmma_f32_16x16x32_bf16(
                        false, fa[im].v, false, fb[in].v, (short)0, c[im][in], false, false);
        }
    };

    const int nsteps = K / BKK;
    stage(0, gA, gB);
    const u16* pa = gA + BKK;
    const u16* pb = gB + BKK;
    for (int step = 0; step < nsteps - 1; ++step) {
        stage((step + 1) & 1, pa, pb);                // prefetch next tile (overlaps compute)
        pa += BKK; pb += BKK;
        // current tile's asyncs are older than the NPEND just issued; per-wave async
        // loads complete in order, so waiting <=NPEND guarantees tile `step&1` landed.
        asm volatile("s_wait_asynccnt %0" :: "i"(NPEND) : "memory");
        __syncthreads();
        compute(step & 1);
        __syncthreads();                              // reads done before buffer re-staged
    }
    asm volatile("s_wait_asynccnt 0" ::: "memory");
    __syncthreads();
    compute((nsteps - 1) & 1);

    // epilogue: C layout — VGPR i: lanes0-15 M=i, lanes16-31 M=8+i; N=lane&15
    #pragma unroll
    for (int im = 0; im < 2; ++im) {
        #pragma unroll
        for (int in = 0; in < NFRAG; ++in) {
            #pragma unroll
            for (int i = 0; i < 8; ++i) {
                int m = m0 + wm * 32 + im * 16 + hlf * 8 + i;
                int n = n0 + wn * (NFRAG * 16) + in * 16 + l15;
                float val = c[im][in][i] + bias[n];
                if (epi == 1)      val = fabsf(val);
                else if (epi == 2) val = 0.5f * val * (1.0f + erff(val * 0.70710678118654752f));
                else if (epi == 3) val += res[(size_t)m * N + n];
                if (outF) outF[(size_t)m * N + n] = val;
                if (outH) outH[(size_t)m * N + n] = f2bf(val);
            }
        }
    }
}

// ---------------- host orchestration ----------------
static inline void launch_gemm(const u16* A, const u16* Bt, const float* bias, const float* res,
                               float* outF, u16* outH, int M, int N, int K, int epi, hipStream_t s) {
    if (N % 128 == 0) {
        dim3 grid(N / 128, M / BM);
        gemm_bf16_kernel<128><<<grid, 256, 0, s>>>(A, Bt, bias, res, outF, outH, M, N, K, epi);
    } else {
        dim3 grid(N / 64, M / BM);
        gemm_bf16_kernel<64><<<grid, 256, 0, s>>>(A, Bt, bias, res, outF, outH, M, N, K, epi);
    }
}

extern "C" void kernel_launch(void* const* d_in, const int* in_sizes, int n_in,
                              void* d_out, int out_size, void* d_ws, size_t ws_size,
                              hipStream_t stream) {
    (void)in_sizes; (void)n_in; (void)out_size; (void)ws_size;
    const float* states = (const float*)d_in[0];
    const float* agent  = (const float*)d_in[1];
    const float* obs    = (const float*)d_in[2];
    const float* Wq = (const float*)d_in[3];  const float* bq = (const float*)d_in[4];
    const float* Wk = (const float*)d_in[5];  const float* bk = (const float*)d_in[6];
    const float* Wv = (const float*)d_in[7];  const float* bv = (const float*)d_in[8];
    const float* wql = (const float*)d_in[9]; const float* wkl = (const float*)d_in[10];
    const float* Wr = (const float*)d_in[11]; const float* br = (const float*)d_in[12];
    const float* Wo = (const float*)d_in[13]; const float* bo = (const float*)d_in[14];
    const float* W1 = (const float*)d_in[15]; const float* b1 = (const float*)d_in[16];
    const float* W2 = (const float*)d_in[17]; const float* b2 = (const float*)d_in[18];
    const float* ln_g = (const float*)d_in[19]; const float* ln_b = (const float*)d_in[20];
    const float* Wl = (const float*)d_in[21]; const float* bl = (const float*)d_in[22];
    float* out = (float*)d_out;

    // workspace carve-out
    char* base = (char*)d_ws;
    size_t off = 0;
    auto alloc = [&](size_t bytes) -> char* {
        char* p = base + off;
        off = (off + bytes + 255) & ~(size_t)255;
        return p;
    };
    const size_t act32 = (size_t)MNROW * DIMD * 4;   // fp32 activation
    const size_t act16 = (size_t)MNROW * DIMD * 2;   // bf16 activation
    u16* sbf   = (u16*)alloc(act16);
    u16* abf   = (u16*)alloc(act16);
    u16* obf   = (u16*)alloc(act16);
    float* qf  = (float*)alloc(act32);
    float* kf  = (float*)alloc(act32);
    float* vf  = (float*)alloc(act32);
    float* gk  = (float*)alloc((size_t)BB * HH * DHD * 4);
    u16* ubf   = (u16*)alloc(act16);
    u16* rbf   = (u16*)alloc(act16);
    float* xf  = (float*)alloc(act32);
    u16* xbf   = (u16*)alloc(act16);
    u16* hbf   = (u16*)alloc((size_t)MNROW * FFD * 2);
    u16* lnbf  = (u16*)alloc(act16);
    u16* wq_t  = (u16*)alloc((size_t)INNR * DIMD * 2);
    u16* wk_t  = (u16*)alloc((size_t)INNR * DIMD * 2);
    u16* wv_t  = (u16*)alloc((size_t)INNR * DIMD * 2);
    u16* wo_t  = (u16*)alloc((size_t)DIMD * INNR * 2);
    u16* wr_t  = (u16*)alloc((size_t)INNR * INNR * 2);
    float* brE = (float*)alloc((size_t)INNR * 4);
    u16* w1_t  = (u16*)alloc((size_t)FFD * DIMD * 2);
    u16* w2_t  = (u16*)alloc((size_t)DIMD * FFD * 2);
    u16* wl_t  = (u16*)alloc((size_t)OUTD * DIMD * 2);

    const size_t actN = (size_t)MNROW * DIMD;
    const int CB = 256;
    // one-time converts
    f32_to_bf16_kernel<<<(unsigned)((actN + CB - 1) / CB), CB, 0, stream>>>(states, sbf, actN);
    f32_to_bf16_kernel<<<(unsigned)((actN + CB - 1) / CB), CB, 0, stream>>>(agent, abf, actN);
    f32_to_bf16_kernel<<<(unsigned)((actN + CB - 1) / CB), CB, 0, stream>>>(obs, obf, actN);
    transpose_bf16_kernel<<<(unsigned)(((size_t)DIMD * OUTD + CB - 1) / CB), CB, 0, stream>>>(Wl, wl_t, DIMD, OUTD);

    for (int l = 0; l < LYR; ++l) {
        const float* Wq_l = Wq + (size_t)l * DIMD * INNR;
        const float* Wk_l = Wk + (size_t)l * DIMD * INNR;
        const float* Wv_l = Wv + (size_t)l * DIMD * INNR;
        const float* Wr_l = Wr + (size_t)l * DHD * DHD;
        const float* Wo_l = Wo + (size_t)l * INNR * DIMD;
        const float* W1_l = W1 + (size_t)l * DIMD * FFD;
        const float* W2_l = W2 + (size_t)l * FFD * DIMD;
        const float* bq_l = bq + (size_t)l * INNR;
        const float* bk_l = bk + (size_t)l * INNR;
        const float* bv_l = bv + (size_t)l * INNR;
        const float* br_l = br + (size_t)l * DHD;
        const float* bo_l = bo + (size_t)l * DIMD;
        const float* b1_l = b1 + (size_t)l * FFD;
        const float* b2_l = b2 + (size_t)l * DIMD;

        size_t wsz = (size_t)DIMD * INNR;
        transpose_bf16_kernel<<<(unsigned)((wsz + CB - 1) / CB), CB, 0, stream>>>(Wq_l, wq_t, DIMD, INNR);
        transpose_bf16_kernel<<<(unsigned)((wsz + CB - 1) / CB), CB, 0, stream>>>(Wk_l, wk_t, DIMD, INNR);
        transpose_bf16_kernel<<<(unsigned)((wsz + CB - 1) / CB), CB, 0, stream>>>(Wv_l, wv_t, DIMD, INNR);
        transpose_bf16_kernel<<<(unsigned)((wsz + CB - 1) / CB), CB, 0, stream>>>(Wo_l, wo_t, INNR, DIMD);
        build_wrexp_kernel<<<(unsigned)(((size_t)INNR * INNR + CB - 1) / CB), CB, 0, stream>>>(Wr_l, br_l, wr_t, brE);
        size_t w1sz = (size_t)DIMD * FFD;
        transpose_bf16_kernel<<<(unsigned)((w1sz + CB - 1) / CB), CB, 0, stream>>>(W1_l, w1_t, DIMD, FFD);
        transpose_bf16_kernel<<<(unsigned)((w1sz + CB - 1) / CB), CB, 0, stream>>>(W2_l, w2_t, FFD, DIMD);

        // q = |states @ Wq + bq|   (fp32 for pooling + residual)
        launch_gemm(sbf, wq_t, bq_l, nullptr, qf, nullptr, MNROW, INNR, DIMD, 1, stream);
        // k = agent @ Wk + bk ; v = obs @ Wv + bv
        launch_gemm(abf, wk_t, bk_l, nullptr, kf, nullptr, MNROW, INNR, DIMD, 0, stream);
        launch_gemm(obf, wv_t, bv_l, nullptr, vf, nullptr, MNROW, INNR, DIMD, 0, stream);
        // pooled global_k per (b,h)
        pool_kernel<<<BB * HH, 256, 0, stream>>>(qf, kf, wql, wkl, gk);
        // u = v * global_k -> bf16
        size_t tot = (size_t)MNROW * INNR;
        gate_kernel<<<(unsigned)((tot + CB - 1) / CB), CB, 0, stream>>>(vf, gk, ubf, tot);
        // r = u @ WrExp + brExp + q  -> bf16
        launch_gemm(ubf, wr_t, brE, qf, nullptr, rbf, MNROW, INNR, INNR, 3, stream);
        // x = r @ Wo + bo  (fp32 + bf16)
        launch_gemm(rbf, wo_t, bo_l, nullptr, xf, xbf, MNROW, DIMD, INNR, 0, stream);
        // h = gelu(x @ W1 + b1) -> bf16 only
        launch_gemm(xbf, w1_t, b1_l, nullptr, nullptr, hbf, MNROW, FFD, DIMD, 2, stream);
        // x = h @ W2 + b2 + x  (in-place residual, fp32)
        launch_gemm(hbf, w2_t, b2_l, xf, xf, nullptr, MNROW, DIMD, FFD, 3, stream);
    }

    // layernorm -> bf16, final projection to d_out
    layernorm_kernel<<<MNROW, 256, 0, stream>>>(xf, ln_g, ln_b, lnbf);
    launch_gemm(lnbf, wl_t, bl, nullptr, out, nullptr, MNROW, OUTD, DIMD, 0, stream);
}
